// ReAGR_26010321945231
// MI455X (gfx1250) — matrix operand order repaired and checked by reading, not compiled, not run on hardware
//
#include <hip/hip_runtime.h>
#include <hip/hip_bf16.h>

#define BB 128
#define SSZ 512
#define EE 128
#define HHD 128
#define HH 256
#define H3 384
#define BS (BB*SSZ) /* 65536 */

typedef __attribute__((ext_vector_type(16))) __bf16 v16bf;
typedef __attribute__((ext_vector_type(8)))  __bf16 v8bf;
typedef __attribute__((ext_vector_type(8)))  float  v8f;

static __device__ __forceinline__ unsigned short f2bf(float f) {
  unsigned int u = __float_as_uint(f);
  u = u + 0x7FFFu + ((u >> 16) & 1u);
  return (unsigned short)(u >> 16);
}
static __device__ __forceinline__ float eluf(float x) { return x > 0.0f ? x : (expf(x) - 1.0f); }
static __device__ __forceinline__ unsigned int rotl32(unsigned int x, int d) { return (x << d) | (x >> (32 - d)); }

// threefry2x32, key = jax.random.key(42) = (0, 42); counters split-half pairing
static __device__ __forceinline__ unsigned int threefry_bit(unsigned int f) {
  bool hi = f >= 65536u;
  unsigned int cnt = hi ? (f - 65536u) : f;
  unsigned int ks0 = 0u, ks1 = 42u, ks2 = 0x1BD11BDAu ^ 0u ^ 42u;
  unsigned int x0 = cnt + ks0, x1 = (cnt + 65536u) + ks1;
  const int ra[4] = {13, 15, 26, 6}, rb[4] = {17, 29, 16, 24};
#define TF4(R) { for (int r = 0; r < 4; ++r) { x0 += x1; x1 = rotl32(x1, R[r]); x1 ^= x0; } }
  TF4(ra); x0 += ks1; x1 += ks2 + 1u;
  TF4(rb); x0 += ks2; x1 += ks0 + 2u;
  TF4(ra); x0 += ks0; x1 += ks1 + 3u;
  TF4(rb); x0 += ks1; x1 += ks2 + 4u;
  TF4(ra); x0 += ks2; x1 += ks0 + 5u;
#undef TF4
  return hi ? x1 : x0;
}
static __device__ __forceinline__ float gumbel_at(unsigned int f) {
  unsigned int bits = threefry_bit(f);
  float u01 = __uint_as_float(0x3f800000u | (bits >> 9)) - 1.0f;
  float u = u01 * (1.0f - 1e-9f) + 1e-9f;
  u = fmaxf(1e-9f, u);
  return -logf(-logf(u));
}

// ---------------- conversion ----------------
__global__ void k_f2bf(const float* __restrict__ s, unsigned short* __restrict__ d, int n) {
  int i = blockIdx.x * 256 + threadIdx.x;
  if (i < n) d[i] = f2bf(s[i]);
}

// ---------------- embedding gather ----------------
__global__ void k_embed(const int* __restrict__ inputs, const float* __restrict__ masks,
                        const float* __restrict__ table, float* __restrict__ raw,
                        unsigned short* __restrict__ embbf) {
  int i = blockIdx.x * 256 + threadIdx.x;
  if (i >= BS * EE) return;
  int bs = i >> 7, e = i & 127;
  int tok = inputs[bs];
  float v = table[(size_t)tok * EE + e];
  raw[i] = v;
  embbf[i] = f2bf(masks[bs] * v);
}

// ------------- generic bf16 WMMA GEMM, 1 M-tile x 4 N-tiles per wave (A reuse) -------------
// out[M,N] = A[M,K] * W[N,K]^T + bias ; N must be a multiple of 64.
__global__ __launch_bounds__(256) void k_gemm_bf16(
    const unsigned short* __restrict__ A, const unsigned short* __restrict__ W,
    const float* __restrict__ bias, float* __restrict__ outF,
    unsigned short* __restrict__ outBF, int M, int K, int N, int act) {
  int gw = (blockIdx.x * 256 + threadIdx.x) >> 5;
  int lane = threadIdx.x & 31;
  int l16 = lane & 15, hf = lane >> 4;
  int ngroups = N >> 6;              // groups of 4 n-tiles
  int mt = gw / ngroups, ng = gw % ngroups;
  int ntbase = ng * 4;
  const __bf16* Ap = (const __bf16*)A + (size_t)(mt * 16 + l16) * K;
  const __bf16* Wp = (const __bf16*)W + (size_t)(ntbase * 16 + l16) * K + hf * 16;
  size_t wstride = (size_t)16 * K;
  v8f acc[4];
  for (int j = 0; j < 4; ++j) acc[j] = (v8f){0.f, 0.f, 0.f, 0.f, 0.f, 0.f, 0.f, 0.f};
  for (int kk = 0; kk < K; kk += 32) {
    union { v16bf v; v8bf h[2]; } au;
    au.h[0] = *(const v8bf*)(Ap + kk + hf * 16);
    au.h[1] = *(const v8bf*)(Ap + kk + 8 + hf * 16);
#pragma unroll
    for (int j = 0; j < 4; ++j) {
      v16bf bv = *(const v16bf*)(Wp + j * wstride + kk);
      acc[j] = __builtin_amdgcn_wmma_f32_16x16x32_bf16(false, au.v, false, bv, (short)0, acc[j], false, false);
    }
  }
#pragma unroll
  for (int j = 0; j < 4; ++j) {
    int col = (ntbase + j) * 16 + l16;
    float bb = bias ? bias[col] : 0.0f;
    for (int r = 0; r < 8; ++r) {
      int orow = mt * 16 + r + hf * 8;
      float v = acc[j][r] + bb;
      if (act == 1) v = eluf(v);
      size_t o = (size_t)orow * N + col;
      if (outF) outF[o] = v;
      if (outBF) outBF[o] = f2bf(v);
    }
  }
}

// ------------- GRU recurrence: one workgroup per direction, h in LDS, WMMA per step -------------
__global__ __launch_bounds__(256) void k_gru_rec(
    const float* __restrict__ xw0, const float* __restrict__ xw1,
    const unsigned short* __restrict__ Whh0, const unsigned short* __restrict__ Whh1,
    const float* __restrict__ bhh0, const float* __restrict__ bhh1,
    float* __restrict__ out) {
  extern __shared__ char smemraw[];
  float* hS = (float*)smemraw;                         // 128*128 f32
  float* gh = hS + BB * HHD;                           // 128*384 f32
  unsigned short* hB = (unsigned short*)(gh + BB * H3);// 128*128 bf16
  int dir = blockIdx.x;
  const float* xw = dir ? xw1 : xw0;
  const unsigned short* Whh = dir ? Whh1 : Whh0;
  const float* bhh = dir ? bhh1 : bhh0;
  int colOff = dir ? HHD : 0;
  int tid = threadIdx.x;
  for (int i = tid; i < BB * HHD; i += 256) { hS[i] = 0.0f; hB[i] = 0; }
  __syncthreads();
  int wave = tid >> 5, lane = tid & 31, hf = lane >> 4, l16 = lane & 15;
  const __bf16* hbp = (const __bf16*)hB + (size_t)(wave * 16 + l16) * HHD;
  const size_t wstride = (size_t)16 * HHD;
  for (int s = 0; s < SSZ; ++s) {
    int ss = dir ? (SSZ - 1 - s) : s;
    // gh = h @ Whh^T  (wave owns 16 batch rows; N=384 as 6 groups of 4 tiles; K=128)
    for (int ntb = 0; ntb < 6; ++ntb) {
      v8f acc[4];
      for (int j = 0; j < 4; ++j) acc[j] = (v8f){0.f, 0.f, 0.f, 0.f, 0.f, 0.f, 0.f, 0.f};
      const __bf16* wp = (const __bf16*)Whh + (size_t)(ntb * 64 + l16) * HHD + hf * 16;
      for (int kk = 0; kk < HHD; kk += 32) {
        union { v16bf v; v8bf h[2]; } au;
        au.h[0] = *(const v8bf*)(hbp + kk + hf * 16);
        au.h[1] = *(const v8bf*)(hbp + kk + 8 + hf * 16);
#pragma unroll
        for (int j = 0; j < 4; ++j) {
          v16bf bv = *(const v16bf*)(wp + j * wstride + kk);
          acc[j] = __builtin_amdgcn_wmma_f32_16x16x32_bf16(false, au.v, false, bv, (short)0, acc[j], false, false);
        }
      }
#pragma unroll
      for (int j = 0; j < 4; ++j) {
        int col = (ntb * 4 + j) * 16 + l16;
        for (int r = 0; r < 8; ++r) gh[(wave * 16 + r + hf * 8) * H3 + col] = acc[j][r];
      }
    }
    __syncthreads();
    // gates
    for (int i = tid; i < BB * HHD; i += 256) {
      int b = i >> 7, j = i & 127;
      size_t xb = ((size_t)(b * SSZ + ss)) * H3;
      float xr = xw[xb + j], xz = xw[xb + HHD + j], xn = xw[xb + 2 * HHD + j];
      float hr = gh[b * H3 + j] + bhh[j];
      float hz = gh[b * H3 + HHD + j] + bhh[HHD + j];
      float hn = gh[b * H3 + 2 * HHD + j] + bhh[2 * HHD + j];
      float rr = 1.0f / (1.0f + expf(-(xr + hr)));
      float zg = 1.0f / (1.0f + expf(-(xz + hz)));
      float nn = tanhf(xn + rr * hn);
      float h2 = (1.0f - zg) * nn + zg * hS[i];
      hS[i] = h2; hB[i] = f2bf(h2);
      out[((size_t)(b * SSZ + ss)) * HH + colOff + j] = h2;
    }
    __syncthreads();
  }
}

// ------------- LayerNorm + gen FC + gumbel-hard (one wave per row) -------------
__global__ __launch_bounds__(256) void k_ln_fc_gumbel(
    const float* __restrict__ X, const float* __restrict__ g, const float* __restrict__ bta,
    const float* __restrict__ fcW, const float* __restrict__ fcb,
    float* __restrict__ genl, float* __restrict__ zout) {
  int wv = threadIdx.x >> 5, lane = threadIdx.x & 31;
  int row = blockIdx.x * 8 + wv;
  const float* x = X + (size_t)row * HH;
  float v[8]; float s = 0.0f;
  for (int t = 0; t < 8; ++t) { v[t] = x[lane + 32 * t]; s += v[t]; }
  for (int m = 1; m < 32; m <<= 1) s += __shfl_xor(s, m);
  float mu = s * (1.0f / HH);
  float q = 0.0f;
  for (int t = 0; t < 8; ++t) { float d = v[t] - mu; q += d * d; }
  for (int m = 1; m < 32; m <<= 1) q += __shfl_xor(q, m);
  float rstd = rsqrtf(q * (1.0f / HH) + 1e-5f);
  float l0 = 0.0f, l1 = 0.0f;
  for (int t = 0; t < 8; ++t) {
    int ch = lane + 32 * t;
    float y = (v[t] - mu) * rstd * g[ch] + bta[ch];
    l0 += y * fcW[ch]; l1 += y * fcW[HH + ch];
  }
  for (int m = 1; m < 32; m <<= 1) { l0 += __shfl_xor(l0, m); l1 += __shfl_xor(l1, m); }
  if (lane == 0) {
    l0 += fcb[0]; l1 += fcb[1];
    genl[row * 2 + 0] = l0; genl[row * 2 + 1] = l1;
    float g0 = gumbel_at((unsigned)(row * 2 + 0));
    float g1 = gumbel_at((unsigned)(row * 2 + 1));
    float p0 = l0 + g0, p1 = l1 + g1;
    float mx = fmaxf(p0, p1);
    float e0 = expf(p0 - mx), e1 = expf(p1 - mx);
    float y0 = e0 / (e0 + e1), y1 = e1 / (e0 + e1);
    int idx = (p1 > p0) ? 1 : 0;
    float h0 = (idx == 0) ? 1.0f : 0.0f, h1 = 1.0f - h0;
    zout[row * 2 + 0] = (h0 - y0) + y0;
    zout[row * 2 + 1] = (h1 - y1) + y1;
  }
}

// ------------- scaled-embedding builder (bf16 A operand) -------------
__global__ void k_scale(const float* __restrict__ raw, const float* __restrict__ masks,
                        const float* __restrict__ z, const int* __restrict__ state,
                        const int* __restrict__ avail, unsigned short* __restrict__ dst, int mode) {
  int i = blockIdx.x * 256 + threadIdx.x;
  if (i >= BS * EE) return;
  int bs = i >> 7;
  float sc;
  if (mode == 0) sc = masks[bs] * z[bs * 2 + 1];
  else if (mode == 3) sc = (float)avail[bs];
  else {
    int zk = (z[bs * 2 + 1] > z[bs * 2 + 0]) ? 1 : 0;
    if (mode == 1) sc = (float)(zk | state[bs]);
    else sc = (float)zk;
  }
  dst[i] = f2bf(sc * raw[i]);
}

// ------------- cls: masked max-pool over S + FC -------------
__global__ __launch_bounds__(256) void k_cls_pool_fc(
    const float* __restrict__ clsout, const float* __restrict__ masks,
    const float* __restrict__ fcW, const float* __restrict__ fcb, float* __restrict__ clsl) {
  __shared__ float pool[256];
  __shared__ float red[256];
  int b = blockIdx.x, tid = threadIdx.x;
  float m = -3.0e38f;
  for (int s = 0; s < SSZ; ++s) {
    float mk = masks[b * SSZ + s];
    float v = clsout[((size_t)(b * SSZ + s)) * HH + tid] * mk + (1.0f - mk) * (-1000000.0f);
    m = fmaxf(m, v);
  }
  pool[tid] = m;
  __syncthreads();
  for (int c = 0; c < 2; ++c) {
    red[tid] = pool[tid] * fcW[c * HH + tid];
    __syncthreads();
    for (int off = 128; off > 0; off >>= 1) { if (tid < off) red[tid] += red[tid + off]; __syncthreads(); }
    if (tid == 0) clsl[b * 2 + c] = red[0] + fcb[c];
    __syncthreads();
  }
}

// ------------- reps = concat(ava_emb, Z_emb - Zk_emb) in bf16 -------------
__global__ void k_reps(const float* __restrict__ ava, const float* __restrict__ Z,
                       const float* __restrict__ Zk, unsigned short* __restrict__ dst) {
  int i = blockIdx.x * 256 + threadIdx.x;
  if (i >= BS * 512) return;
  int bs = i >> 9, d = i & 511;
  float v = (d < HH) ? ava[(size_t)bs * HH + d] : (Z[(size_t)bs * HH + d - HH] - Zk[(size_t)bs * HH + d - HH]);
  dst[i] = f2bf(v);
}

// ------------- h2 GEMM fused with W3 contraction (deterministic, shfl reduce) -------------
__global__ __launch_bounds__(256) void k_h2_fused(
    const unsigned short* __restrict__ h1bf, const unsigned short* __restrict__ W2bf,
    const float* __restrict__ b2, const float* __restrict__ W3, const float* __restrict__ b3,
    float* __restrict__ aprob) {
  int gw = (blockIdx.x * 256 + threadIdx.x) >> 5;
  int lane = threadIdx.x & 31, l16 = lane & 15, hf = lane >> 4;
  int mt = gw >> 1, c = gw & 1;
  const __bf16* Ap = (const __bf16*)h1bf + (size_t)c * BS * 512 + (size_t)(mt * 16 + l16) * 512;
  const size_t wstride = (size_t)16 * 512;
  float p[8] = {0.f, 0.f, 0.f, 0.f, 0.f, 0.f, 0.f, 0.f};
  for (int ntb = 0; ntb < 4; ++ntb) {   // 4 groups of 4 n-tiles -> N=256
    v8f acc[4];
    for (int j = 0; j < 4; ++j) acc[j] = (v8f){0.f, 0.f, 0.f, 0.f, 0.f, 0.f, 0.f, 0.f};
    const __bf16* wp = (const __bf16*)W2bf + ((size_t)c * HH + ntb * 64 + l16) * 512 + hf * 16;
    for (int kk = 0; kk < 512; kk += 32) {
      union { v16bf v; v8bf h[2]; } au;
      au.h[0] = *(const v8bf*)(Ap + kk + hf * 16);
      au.h[1] = *(const v8bf*)(Ap + kk + 8 + hf * 16);
#pragma unroll
      for (int j = 0; j < 4; ++j) {
        v16bf bv = *(const v16bf*)(wp + j * wstride + kk);
        acc[j] = __builtin_amdgcn_wmma_f32_16x16x32_bf16(false, au.v, false, bv, (short)0, acc[j], false, false);
      }
    }
#pragma unroll
    for (int j = 0; j < 4; ++j) {
      int col = (ntb * 4 + j) * 16 + l16;
      float bb = b2[c * HH + col], w3 = W3[c * HH + col];
      for (int r = 0; r < 8; ++r) p[r] += eluf(acc[j][r] + bb) * w3;
    }
  }
  for (int r = 0; r < 8; ++r)
    for (int m = 1; m < 16; m <<= 1) p[r] += __shfl_xor(p[r], m);
  if (l16 == 0) {
    for (int r = 0; r < 8; ++r) {
      int orow = mt * 16 + r + hf * 8;
      aprob[(size_t)orow * 2 + c] = p[r] + b3[c];
    }
  }
}

// ------------- per-batch: gather label class, softmax over S, top-3 -------------
__global__ __launch_bounds__(256) void k_ava_topk(
    const float* __restrict__ aprob, const int* __restrict__ labels,
    float* __restrict__ avaout, float* __restrict__ addp, float* __restrict__ adda) {
  __shared__ float p[SSZ];
  __shared__ float rv[256];
  __shared__ int ri[256];
  int b = blockIdx.x, tid = threadIdx.x;
  int lab = labels[b]; lab = lab < 0 ? 0 : (lab > 1 ? 1 : lab);
  for (int s = tid; s < SSZ; s += 256) {
    float v = aprob[((size_t)(b * SSZ + s)) * 2 + lab];
    p[s] = v;
    avaout[b * SSZ + s] = v;
  }
  __syncthreads();
  float lm = -3.0e38f;
  for (int s = tid; s < SSZ; s += 256) lm = fmaxf(lm, p[s]);
  rv[tid] = lm; __syncthreads();
  for (int off = 128; off > 0; off >>= 1) { if (tid < off) rv[tid] = fmaxf(rv[tid], rv[tid + off]); __syncthreads(); }
  float mx = rv[0]; __syncthreads();
  float ls = 0.0f;
  for (int s = tid; s < SSZ; s += 256) ls += expf(p[s] - mx);
  rv[tid] = ls; __syncthreads();
  for (int off = 128; off > 0; off >>= 1) { if (tid < off) rv[tid] += rv[tid + off]; __syncthreads(); }
  float denom = rv[0]; __syncthreads();
  for (int s = tid; s < SSZ; s += 256) p[s] = expf(p[s] - mx) / denom;
  __syncthreads();
  for (int t = 0; t < 3; ++t) {
    float bv = -3.0e38f; int bi = 1 << 20;
    for (int s = tid; s < SSZ; s += 256)
      if (p[s] > bv || (p[s] == bv && s < bi)) { bv = p[s]; bi = s; }
    rv[tid] = bv; ri[tid] = bi; __syncthreads();
    for (int off = 128; off > 0; off >>= 1) {
      if (tid < off) {
        if (rv[tid + off] > rv[tid] || (rv[tid + off] == rv[tid] && ri[tid + off] < ri[tid])) {
          rv[tid] = rv[tid + off]; ri[tid] = ri[tid + off];
        }
      }
      __syncthreads();
    }
    if (tid == 0) {
      addp[b * 3 + t] = rv[0];
      adda[b * 3 + t] = (float)ri[0];
      p[ri[0]] = -3.0e38f;
    }
    __syncthreads();
  }
}

// ================= host =================
static const size_t OFF_RAW    = 0ull;
static const size_t OFF_EMB_BF = 33554432ull;
static const size_t OFF_XWF    = 50331648ull;
static const size_t OFF_XWR    = 150994944ull;
static const size_t OFF_OUTGEN = 251658240ull;
static const size_t OFF_OUTCLS = 318767104ull;
static const size_t OFF_ZEMB   = 385875968ull;
static const size_t OFF_ZKEMB  = 452984832ull;
static const size_t OFF_AVAEMB = 520093696ull;
static const size_t OFF_SCL    = 587202560ull;
static const size_t OFF_REPS   = 603979776ull;
static const size_t OFF_ACT    = 671088640ull;
static const size_t OFF_WBF    = 671612928ull;
static const size_t OFF_H1BF   = OFF_XWF; // alias over dead xw buffers (134MB <= 201MB)

extern "C" void kernel_launch(void* const* d_in, const int* in_sizes, int n_in,
                              void* d_out, int out_size, void* d_ws, size_t ws_size,
                              hipStream_t stream) {
  (void)in_sizes; (void)out_size; (void)ws_size;
  char* ws = (char*)d_ws;
  float* out = (float*)d_out;

  const int* p_inputs = (const int*)d_in[0];
  const float* p_masks = (const float*)d_in[1];
  const int* p_state = (const int*)d_in[2];
  const int* p_avail = (const int*)d_in[3];
  const int* p_labels = (const int*)d_in[4];
  const float* p_embT = (const float*)d_in[6];
  const float *gen[8], *cls[8], *enc[8];
  const float *ln_g, *ln_b, *gfcW, *gfcb, *cfcW, *cfcb, *W1, *b1, *W2, *b2, *W3, *b3;
  if (n_in >= 40) { // tuples flattened to leaves
    for (int i = 0; i < 8; ++i) {
      gen[i] = (const float*)d_in[7 + i];
      cls[i] = (const float*)d_in[15 + i];
      enc[i] = (const float*)d_in[23 + i];
    }
    ln_g = (const float*)d_in[31]; ln_b = (const float*)d_in[32];
    gfcW = (const float*)d_in[33]; gfcb = (const float*)d_in[34];
    cfcW = (const float*)d_in[35]; cfcb = (const float*)d_in[36];
    W1 = (const float*)d_in[37]; b1 = (const float*)d_in[38];
    W2 = (const float*)d_in[39]; b2 = (const float*)d_in[40];
    W3 = (const float*)d_in[41]; b3 = (const float*)d_in[42];
  } else { // tuples concatenated per dict key
    static const size_t off8[8] = {0, 49152, 98304, 98688, 99072, 148224, 197376, 197760};
    const float* gb = (const float*)d_in[7];
    const float* cb = (const float*)d_in[8];
    const float* eb = (const float*)d_in[9];
    for (int i = 0; i < 8; ++i) { gen[i] = gb + off8[i]; cls[i] = cb + off8[i]; enc[i] = eb + off8[i]; }
    ln_g = (const float*)d_in[10]; ln_b = (const float*)d_in[11];
    gfcW = (const float*)d_in[12]; gfcb = (const float*)d_in[13];
    cfcW = (const float*)d_in[14]; cfcb = (const float*)d_in[15];
    W1 = (const float*)d_in[16]; b1 = (const float*)d_in[17];
    W2 = (const float*)d_in[18]; b2 = (const float*)d_in[19];
    W3 = (const float*)d_in[20]; b3 = (const float*)d_in[21];
  }

  float* raw = (float*)(ws + OFF_RAW);
  unsigned short* embbf = (unsigned short*)(ws + OFF_EMB_BF);
  float* xwf = (float*)(ws + OFF_XWF);
  float* xwr = (float*)(ws + OFF_XWR);
  float* outgen = (float*)(ws + OFF_OUTGEN);
  float* outcls = (float*)(ws + OFF_OUTCLS);
  float* Zemb = (float*)(ws + OFF_ZEMB);
  float* Zkemb = (float*)(ws + OFF_ZKEMB);
  float* avaemb = (float*)(ws + OFF_AVAEMB);
  unsigned short* sclbf = (unsigned short*)(ws + OFF_SCL);
  unsigned short* repsbf = (unsigned short*)(ws + OFF_REPS);
  float* aprob = (float*)(ws + OFF_ACT);
  unsigned short* wbf = (unsigned short*)(ws + OFF_WBF);
  unsigned short* h1bf = (unsigned short*)(ws + OFF_H1BF);

  // bf16 weight sub-offsets (elements)
  unsigned short* gWihF = wbf + 0;      unsigned short* gWhhF = wbf + 49152;
  unsigned short* gWihR = wbf + 98304;  unsigned short* gWhhR = wbf + 147456;
  unsigned short* cWihF = wbf + 196608; unsigned short* cWhhF = wbf + 245760;
  unsigned short* cWihR = wbf + 294912; unsigned short* cWhhR = wbf + 344064;
  unsigned short* eWihF = wbf + 393216; unsigned short* eWhhF = wbf + 442368;
  unsigned short* eWihR = wbf + 491520; unsigned short* eWhhR = wbf + 540672;
  unsigned short* W1bf = wbf + 589824;  unsigned short* W2bf = wbf + 1114112;

  float* o_genl = out;
  float* o_z = out + 131072;
  float* o_clsl = out + 262144;
  float* o_ava = out + 262400;
  float* o_addp = out + 327936;
  float* o_adda = out + 328320;

  auto conv = [&](const float* s, unsigned short* d, int n) {
    k_f2bf<<<(n + 255) / 256, 256, 0, stream>>>(s, d, n);
  };
  conv(gen[0], gWihF, 49152); conv(gen[1], gWhhF, 49152);
  conv(gen[4], gWihR, 49152); conv(gen[5], gWhhR, 49152);
  conv(cls[0], cWihF, 49152); conv(cls[1], cWhhF, 49152);
  conv(cls[4], cWihR, 49152); conv(cls[5], cWhhR, 49152);
  conv(enc[0], eWihF, 49152); conv(enc[1], eWhhF, 49152);
  conv(enc[4], eWihR, 49152); conv(enc[5], eWhhR, 49152);
  conv(W1, W1bf, 524288); conv(W2, W2bf, 262144);

  k_embed<<<(BS * EE) / 256, 256, 0, stream>>>(p_inputs, p_masks, p_embT, raw, embbf);

  const int xw_blocks = (BS / 16) * (H3 / 64) / 8; // 3072 (4 n-tiles per wave)
  const size_t rec_lds = (size_t)BB * HHD * 4 + (size_t)BB * H3 * 4 + (size_t)BB * HHD * 2; // 294912

  auto run_bigru = [&](const unsigned short* Abf,
                       const unsigned short* wihF, const unsigned short* wihR,
                       const unsigned short* whhF, const unsigned short* whhR,
                       const float* bihF, const float* bihR,
                       const float* bhhF, const float* bhhR, float* outp) {
    k_gemm_bf16<<<xw_blocks, 256, 0, stream>>>(Abf, wihF, bihF, xwf, nullptr, BS, EE, H3, 0);
    k_gemm_bf16<<<xw_blocks, 256, 0, stream>>>(Abf, wihR, bihR, xwr, nullptr, BS, EE, H3, 0);
    k_gru_rec<<<2, 256, rec_lds, stream>>>(xwf, xwr, whhF, whhR, bhhF, bhhR, outp);
  };

  // gen bigru -> layernorm/fc/gumbel -> z
  run_bigru(embbf, gWihF, gWihR, gWhhF, gWhhR, gen[2], gen[6], gen[3], gen[7], outgen);
  k_ln_fc_gumbel<<<BS / 8, 256, 0, stream>>>(outgen, ln_g, ln_b, gfcW, gfcb, o_genl, o_z);

  // cls bigru on emb * z1 (masked)
  k_scale<<<(BS * EE) / 256, 256, 0, stream>>>(raw, p_masks, o_z, p_state, p_avail, sclbf, 0);
  run_bigru(sclbf, cWihF, cWihR, cWhhF, cWhhR, cls[2], cls[6], cls[3], cls[7], outcls);
  k_cls_pool_fc<<<BB, 256, 0, stream>>>(outcls, p_masks, cfcW, cfcb, o_clsl);

  // enc bigru streams: Zm, zk, ava
  k_scale<<<(BS * EE) / 256, 256, 0, stream>>>(raw, p_masks, o_z, p_state, p_avail, sclbf, 1);
  run_bigru(sclbf, eWihF, eWihR, eWhhF, eWhhR, enc[2], enc[6], enc[3], enc[7], Zemb);
  k_scale<<<(BS * EE) / 256, 256, 0, stream>>>(raw, p_masks, o_z, p_state, p_avail, sclbf, 2);
  run_bigru(sclbf, eWihF, eWihR, eWhhF, eWhhR, enc[2], enc[6], enc[3], enc[7], Zkemb);
  k_scale<<<(BS * EE) / 256, 256, 0, stream>>>(raw, p_masks, o_z, p_state, p_avail, sclbf, 3);
  run_bigru(sclbf, eWihF, eWihR, eWhhF, eWhhR, enc[2], enc[6], enc[3], enc[7], avaemb);

  // reps + explanation MLP
  k_reps<<<(BS * 512) / 256, 256, 0, stream>>>(avaemb, Zemb, Zkemb, repsbf);
  const int h1_blocks = (BS / 16) * (512 / 64) / 8; // 4096 (4 n-tiles per wave)
  for (int c = 0; c < 2; ++c) {
    k_gemm_bf16<<<h1_blocks, 256, 0, stream>>>(
        repsbf, W1bf + (size_t)c * 262144, b1 + c * 512, nullptr,
        h1bf + (size_t)c * BS * 512, BS, 512, 512, 1);
  }
  k_h2_fused<<<(BS / 16) * 2 / 8, 256, 0, stream>>>(h1bf, W2bf, b2, W3, b3, aprob);

  k_ava_topk<<<BB, 256, 0, stream>>>(aprob, p_labels, o_ava, o_addp, o_adda);
}
